// DinomalyAnomalyHead_27023934227272
// MI455X (gfx1250) — matrix operand order repaired and checked by reading, not compiled, not run on hardware
//
#include <hip/hip_runtime.h>
#include <hip/hip_bf16.h>

// ---------------------------------------------------------------------------
// Types for CDNA5 WMMA / async-LDS
// ---------------------------------------------------------------------------
typedef __attribute__((ext_vector_type(16))) __bf16 v16bf;
typedef __attribute__((ext_vector_type(8)))  float  v8f;
typedef __attribute__((ext_vector_type(4)))  int    v4i;

#define AS_GLOBAL __attribute__((address_space(1)))
#define AS_LOCAL  __attribute__((address_space(3)))

#define TILE_M 128
#define TILE_N 256
#define TILE_K 32
#define LDS_A_STRIDE 34   // bf16 elems per As row (32 + pad) -> 68B, odd DWORD stride
#define LDS_B_STRIDE 34   // bf16 elems per Bs row (Bs stored [n][k])

#if defined(__gfx1250__)
#if __has_builtin(__builtin_amdgcn_global_load_async_to_lds_b128)
#define ASYNC_LDS_COPY 1
#endif
#endif

// ---------------------------------------------------------------------------
// Generic bf16-WMMA GEMM: out = epilogue(A[MxK] * W[KxN] + bias)
// Template epilogue: exact GELU / residual add / accumulator += (straight-line).
// M multiple of 128, N multiple of 256, K multiple of 32.
// Block = 256 threads = 8 waves in 2x4 grid; each wave owns a 64x64 sub-tile
// -> 16 v_wmma_f32_16x16x32_bf16 per K-step per wave.
// ---------------------------------------------------------------------------
template<bool GELU, bool RES, bool ACC>
__global__ __launch_bounds__(256)
void gemm_bf16_wmma(const float* __restrict__ A, const float* __restrict__ W,
                    const float* __restrict__ bias, const float* __restrict__ resid,
                    float* __restrict__ out, float* __restrict__ accum,
                    int M, int N, int K)
{
    __shared__ __bf16 As[TILE_M * LDS_A_STRIDE];  // [m][k]
    __shared__ __bf16 Bs[TILE_N * LDS_B_STRIDE];  // [n][k] (transposed from KxN)

    const int tid  = threadIdx.x;
    const int lane = tid & 31;
    const int wave = tid >> 5;       // 0..7
    const int wr   = wave >> 2;      // 0..1 : 64-row group
    const int wc   = wave & 3;       // 0..3 : 64-col group
    const int l16  = lane & 15;
    const int hi   = lane >> 4;      // lane half

    const int mtile = blockIdx.y * TILE_M;
    const int ntile = blockIdx.x * TILE_N;

    const v8f vzero = {0.f,0.f,0.f,0.f,0.f,0.f,0.f,0.f};
    v8f acc[4][4];
    #pragma unroll
    for (int i = 0; i < 4; i++)
        #pragma unroll
        for (int j = 0; j < 4; j++) acc[i][j] = vzero;

    // staging index precomputes
    const int a_kq    = (tid & 7) * 4;       // 0..28
    const int a_rbase = tid >> 3;            // 0..31
    const int b_ng    = (tid & 63) * 4;      // 0..252
    const int b_kb    = tid >> 6;            // 0..3

    for (int k0 = 0; k0 < K; k0 += TILE_K) {
        // ---- stage A tile 128x32 fp32 -> bf16 LDS (float4/thread, 4 passes)
        #pragma unroll
        for (int p = 0; p < 4; p++) {
            const int row = p * 32 + a_rbase;
            const float4 v = *(const float4*)(A + (size_t)(mtile + row) * K + k0 + a_kq);
            __bf16* dst = &As[row * LDS_A_STRIDE + a_kq];
            dst[0] = (__bf16)v.x; dst[1] = (__bf16)v.y;
            dst[2] = (__bf16)v.z; dst[3] = (__bf16)v.w;
        }
        // ---- stage B tile 32x256 fp32 -> bf16 LDS transposed to [n][k] (8 passes)
        #pragma unroll
        for (int p = 0; p < 8; p++) {
            const int kk = p * 4 + b_kb;     // 0..31
            const float4 v = *(const float4*)(W + (size_t)(k0 + kk) * N + ntile + b_ng);
            Bs[(b_ng + 0) * LDS_B_STRIDE + kk] = (__bf16)v.x;
            Bs[(b_ng + 1) * LDS_B_STRIDE + kk] = (__bf16)v.y;
            Bs[(b_ng + 2) * LDS_B_STRIDE + kk] = (__bf16)v.z;
            Bs[(b_ng + 3) * LDS_B_STRIDE + kk] = (__bf16)v.w;
        }
        // hint next K-tile into cache while we compute this one
        if (k0 + TILE_K < K) {
            __builtin_prefetch(A + (size_t)(mtile + a_rbase) * K + k0 + TILE_K + a_kq, 0, 0);
            __builtin_prefetch(W + (size_t)(k0 + TILE_K + b_kb) * N + ntile + b_ng, 0, 0);
        }
        __syncthreads();

        // ---- gather fragments per ISA 16-bit A/B VGPR layouts
        v16bf afrag[4];
        #pragma unroll
        for (int mi = 0; mi < 4; mi++) {
            const int m = wr * 64 + mi * 16 + l16;
            const __bf16* ap = &As[m * LDS_A_STRIDE];
            v16bf a;
            #pragma unroll
            for (int j = 0; j < 8; j++) {
                const int kq = (j >> 2) * 16 + hi * 8 + (j & 3) * 2;
                a[2 * j]     = ap[kq];
                a[2 * j + 1] = ap[kq + 1];
            }
            afrag[mi] = a;
        }
        v16bf bfrag[4];
        #pragma unroll
        for (int ni = 0; ni < 4; ni++) {
            const int n = wc * 64 + ni * 16 + l16;
            const __bf16* bp = &Bs[n * LDS_B_STRIDE];
            v16bf bfv;
            #pragma unroll
            for (int j = 0; j < 8; j++) {
                const int kq = (j >> 2) * 16 + hi * 8 + (j & 3) * 2;
                bfv[2 * j]     = bp[kq];
                bfv[2 * j + 1] = bp[kq + 1];
            }
            bfrag[ni] = bfv;
        }

        #pragma unroll
        for (int mi = 0; mi < 4; mi++)
            #pragma unroll
            for (int ni = 0; ni < 4; ni++) {
#if defined(__gfx1250__)
                acc[mi][ni] = __builtin_amdgcn_wmma_f32_16x16x32_bf16(
                    false, afrag[mi], false, bfrag[ni],
                    (short)0, acc[mi][ni], false, false);
#endif
            }
        __syncthreads();
    }

    // ---- epilogue: C/D layout => row = base + 8*hi + r, col = base + l16
    #pragma unroll
    for (int mi = 0; mi < 4; mi++) {
        #pragma unroll
        for (int ni = 0; ni < 4; ni++) {
            const int col = ntile + wc * 64 + ni * 16 + l16;
            const float bv = bias[col];
            #pragma unroll
            for (int r = 0; r < 8; r++) {
                const int row = mtile + wr * 64 + mi * 16 + hi * 8 + r;
                float v = acc[mi][ni][r] + bv;
                if (GELU) v = 0.5f * v * (1.0f + erff(v * 0.70710678118654752f));
                const size_t idx = (size_t)row * N + col;
                if (RES) v += resid[idx];
                out[idx] = v;
                if (ACC) accum[idx] += v;
            }
        }
    }
}

// ---------------------------------------------------------------------------
// x[b,n,c] = mean over L=8 of en_feats[b,l,n,c]
// ---------------------------------------------------------------------------
__global__ void mean8_kernel(const float* __restrict__ en, float* __restrict__ x, int total)
{
    const int i = blockIdx.x * blockDim.x + threadIdx.x;
    if (i >= total) return;
    const int b = i >> 20;               // N*C = 1M
    const int r = i & ((1 << 20) - 1);
    const float* p = en + ((size_t)b << 23) + r;   // b * L*N*C
    float s = 0.0f;
    #pragma unroll
    for (int l = 0; l < 8; l++) s += p[(size_t)l << 20];
    x[i] = s * 0.125f;
}

// ---------------------------------------------------------------------------
// LayerNorm over C=1024; one 256-thread block per row (4096 rows)
// ---------------------------------------------------------------------------
__global__ __launch_bounds__(256)
void layernorm_kernel(const float* __restrict__ x, const float* __restrict__ w,
                      const float* __restrict__ b, float* __restrict__ y)
{
    __shared__ float s1[256], s2[256];
    const int row = blockIdx.x;
    const int tid = threadIdx.x;
    const float* xr = x + (size_t)row * 1024;
    float lv[4], sum = 0.0f, sq = 0.0f;
    #pragma unroll
    for (int i = 0; i < 4; i++) {
        const float v = xr[tid + i * 256];
        lv[i] = v; sum += v; sq += v * v;
    }
    s1[tid] = sum; s2[tid] = sq;
    __syncthreads();
    for (int off = 128; off > 0; off >>= 1) {
        if (tid < off) { s1[tid] += s1[tid + off]; s2[tid] += s2[tid + off]; }
        __syncthreads();
    }
    const float mean = s1[0] * (1.0f / 1024.0f);
    const float var  = s2[0] * (1.0f / 1024.0f) - mean * mean;
    const float rstd = rsqrtf(var + 1e-8f);
    float* yr = y + (size_t)row * 1024;
    #pragma unroll
    for (int i = 0; i < 4; i++) {
        const int c = tid + i * 256;
        yr[c] = (lv[i] - mean) * rstd * w[c] + b[c];
    }
}

// ---------------------------------------------------------------------------
// Linear attention stage 1: kv[bh] = sum_s (elu(k)+1) outer v ; ksum[bh]
// One block per (b,h); qkv rows are [q(1024) | k(1024) | v(1024)]
// ---------------------------------------------------------------------------
__global__ __launch_bounds__(256)
void attn_kv_kernel(const float* __restrict__ qkv, float* __restrict__ kv,
                    float* __restrict__ ksum)
{
    __shared__ float ks[4][64], vs[4][64];
    const int bh = blockIdx.x;                 // 0..63
    const int b = bh >> 4, h = bh & 15;
    const int tid = threadIdx.x;
    const int td = tid >> 4, te = tid & 15;    // thread owns 4x4 cell block
    float acc[4][4] = {};
    float ksl = 0.0f;
    const float* base = qkv + (size_t)b * 1024 * 3072;
    const int kcol = 1024 + h * 64;
    const int vcol = 2048 + h * 64;
    const int sl = tid >> 6, e = tid & 63;     // staging assignment

    for (int s0 = 0; s0 < 1024; s0 += 4) {
        const float* rowp = base + (size_t)(s0 + sl) * 3072;
        const float kraw = rowp[kcol + e];
        ks[sl][e] = kraw > 0.0f ? kraw + 1.0f : expf(kraw);   // elu+1
        vs[sl][e] = rowp[vcol + e];
        __syncthreads();
        #pragma unroll
        for (int s = 0; s < 4; s++) {
            float kd[4], ve[4];
            #pragma unroll
            for (int i = 0; i < 4; i++) { kd[i] = ks[s][td * 4 + i]; ve[i] = vs[s][te * 4 + i]; }
            #pragma unroll
            for (int i = 0; i < 4; i++)
                #pragma unroll
                for (int j = 0; j < 4; j++) acc[i][j] += kd[i] * ve[j];
            if (tid < 64) ksl += ks[s][tid];
        }
        __syncthreads();
    }
    float* kvp = kv + (size_t)bh * 4096;
    #pragma unroll
    for (int i = 0; i < 4; i++)
        #pragma unroll
        for (int j = 0; j < 4; j++)
            kvp[(td * 4 + i) * 64 + te * 4 + j] = acc[i][j];
    if (tid < 64) ksum[bh * 64 + tid] = ksl;
}

// ---------------------------------------------------------------------------
// Linear attention stage 2: out[b,s,h*64+e] = (q . kv[:,e]) / (q . ksum)
// q = elu(raw)+1. One block per (b,h); kv staged in LDS via CDNA5 async copy.
// ---------------------------------------------------------------------------
__global__ __launch_bounds__(256)
void attn_out_kernel(const float* __restrict__ qkv, const float* __restrict__ kv,
                     const float* __restrict__ ksum, float* __restrict__ out)
{
    __shared__ float kvs[64 * 64];
    __shared__ float kss[64];
    __shared__ float qs[4][64];
    const int bh = blockIdx.x;
    const int b = bh >> 4, h = bh & 15;
    const int tid = threadIdx.x;
    const float* kvp = kv + (size_t)bh * 4096;

    // stage kv (16KB, raw copy: perfect for async global->LDS DMA)
#if ASYNC_LDS_COPY
    #pragma unroll
    for (int p = 0; p < 4; p++) {
        __builtin_amdgcn_global_load_async_to_lds_b128(
            (AS_GLOBAL v4i*)(kvp + p * 1024 + tid * 4),
            (AS_LOCAL  v4i*)(&kvs[p * 1024 + tid * 4]),
            0, 0);
    }
#if __has_builtin(__builtin_amdgcn_s_wait_asynccnt)
    __builtin_amdgcn_s_wait_asynccnt(0);
#else
    asm volatile("s_wait_asynccnt 0" ::: "memory");
#endif
#else
    for (int i = tid; i < 4096; i += 256) kvs[i] = kvp[i];
#endif
    if (tid < 64) kss[tid] = ksum[bh * 64 + tid];
    __syncthreads();

    const float* base = qkv + (size_t)b * 1024 * 3072;
    const int qcol = h * 64;
    const int sl = tid >> 6, e = tid & 63;
    float* ob = out + (size_t)b * 1024 * 1024 + qcol;

    for (int s0 = 0; s0 < 1024; s0 += 4) {
        const float qraw = base[(size_t)(s0 + sl) * 3072 + qcol + e];
        qs[sl][e] = qraw > 0.0f ? qraw + 1.0f : expf(qraw);
        __syncthreads();
        float dot = 0.0f, den = 0.0f;
        #pragma unroll 8
        for (int d = 0; d < 64; d++) {
            const float q = qs[sl][d];
            dot += q * kvs[d * 64 + e];
            den += q * kss[d];
        }
        ob[(size_t)(s0 + sl) * 1024 + e] = dot / den;
        __syncthreads();
    }
}

// ---------------------------------------------------------------------------
// Fused layer-mean + transpose: dst[b][c][n] = scale * sum_l src[...][n][c]
// 32x32 LDS tile. N = C = 1024. grid = (32, 32, B)
// ---------------------------------------------------------------------------
__global__ __launch_bounds__(256)
void transpose_out_kernel(const float* __restrict__ src, float* __restrict__ dst,
                          size_t bStride, int lCount, size_t lStride, float scale)
{
    __shared__ float tile[32][33];
    const int b = blockIdx.z;
    const int nbase = blockIdx.x * 32;
    const int cbase = blockIdx.y * 32;
    const int tc = threadIdx.x & 31, tr = threadIdx.x >> 5;  // 8 rows per pass
    const float* sb = src + (size_t)b * bStride;
    #pragma unroll
    for (int rr = 0; rr < 4; rr++) {
        const int n = nbase + tr + rr * 8;
        float s = 0.0f;
        for (int l = 0; l < lCount; l++)
            s += sb[(size_t)l * lStride + (size_t)n * 1024 + cbase + tc];
        tile[tr + rr * 8][tc] = s;
    }
    __syncthreads();
    #pragma unroll
    for (int rr = 0; rr < 4; rr++) {
        const int c = cbase + tr + rr * 8;
        dst[((size_t)b * 1024 + c) * 1024 + nbase + tc] = tile[tc][tr + rr * 8] * scale;
    }
}

__global__ void zero_kernel(float* __restrict__ p, int n)
{
    const int i = blockIdx.x * blockDim.x + threadIdx.x;
    if (i < n) p[i] = 0.0f;
}

// ---------------------------------------------------------------------------
// Host-side orchestration
// ---------------------------------------------------------------------------
static inline void launch_gemm(const float* A, const float* W, const float* bias,
                               const float* resid, float* out, float* accum,
                               int M, int N, int K, int gelu, hipStream_t s)
{
    const dim3 g(N / TILE_N, M / TILE_M), b(256);
    if (accum)      gemm_bf16_wmma<false, true , true ><<<g, b, 0, s>>>(A, W, bias, resid, out, accum, M, N, K);
    else if (resid) gemm_bf16_wmma<false, true , false><<<g, b, 0, s>>>(A, W, bias, resid, out, accum, M, N, K);
    else if (gelu)  gemm_bf16_wmma<true , false, false><<<g, b, 0, s>>>(A, W, bias, resid, out, accum, M, N, K);
    else            gemm_bf16_wmma<false, false, false><<<g, b, 0, s>>>(A, W, bias, resid, out, accum, M, N, K);
}

extern "C" void kernel_launch(void* const* d_in, const int* in_sizes, int n_in,
                              void* d_out, int out_size, void* d_ws, size_t ws_size,
                              hipStream_t stream)
{
    (void)in_sizes; (void)n_in; (void)out_size; (void)ws_size;
    const float* en        = (const float*)d_in[0];
    const float* bn_fc1_w  = (const float*)d_in[1];
    const float* bn_fc1_b  = (const float*)d_in[2];
    const float* bn_fc2_w  = (const float*)d_in[3];
    const float* bn_fc2_b  = (const float*)d_in[4];
    const float* ln1_w     = (const float*)d_in[5];
    const float* ln1_b     = (const float*)d_in[6];
    const float* qkv_w     = (const float*)d_in[7];
    const float* qkv_b     = (const float*)d_in[8];
    const float* proj_w    = (const float*)d_in[9];
    const float* proj_b    = (const float*)d_in[10];
    const float* ln2_w     = (const float*)d_in[11];
    const float* ln2_b     = (const float*)d_in[12];
    const float* f1w       = (const float*)d_in[13];
    const float* f1b       = (const float*)d_in[14];
    const float* f2w       = (const float*)d_in[15];
    const float* f2b       = (const float*)d_in[16];

    float* ws    = (float*)d_ws;
    float* carry = ws;                            //  4M floats
    float* tmp1  = ws + ((size_t)4  << 20);       //  4M
    float* big   = ws + ((size_t)8  << 20);       // 16M (qkv 12M / hid 16M)
    float* acc0  = ws + ((size_t)24 << 20);       //  4M
    float* acc1  = ws + ((size_t)28 << 20);       //  4M
    float* kvb   = big + ((size_t)12 << 20);      // 64*4096, inside big after qkv
    float* ksb   = kvb + 64 * 4096;               // 64*64
    float* outp  = (float*)d_out;

    const int M = 4096, C = 1024, HID = 4096, C3 = 3072;
    const dim3 blk(256);

    // zero de0/de1 accumulators (acc0, acc1 contiguous: 8M floats)
    zero_kernel<<<dim3((8u << 20) / 256), blk, 0, stream>>>(acc0, 8 << 20);

    // x = mean over L
    mean8_kernel<<<dim3((4u << 20) / 256), blk, 0, stream>>>(en, carry, 4 << 20);

    // bottleneck MLP
    launch_gemm(carry, bn_fc1_w, bn_fc1_b, nullptr, big,   nullptr, M, HID, C,  1, stream);
    launch_gemm(big,   bn_fc2_w, bn_fc2_b, nullptr, carry, nullptr, M, C,  HID, 0, stream);

    // transformer blocks
    for (int d = 0; d < 8; d++) {
        layernorm_kernel<<<dim3(M), blk, 0, stream>>>(
            carry, ln1_w + d * C, ln1_b + d * C, tmp1);
        launch_gemm(tmp1, qkv_w + (size_t)d * C * C3, qkv_b + d * C3,
                    nullptr, big, nullptr, M, C3, C, 0, stream);
        attn_kv_kernel<<<dim3(64), blk, 0, stream>>>(big, kvb, ksb);
        attn_out_kernel<<<dim3(64), blk, 0, stream>>>(big, kvb, ksb, tmp1);
        launch_gemm(tmp1, proj_w + (size_t)d * C * C, proj_b + d * C,
                    carry, carry, nullptr, M, C, C, 0, stream);
        layernorm_kernel<<<dim3(M), blk, 0, stream>>>(
            carry, ln2_w + d * C, ln2_b + d * C, tmp1);
        launch_gemm(tmp1, f1w + (size_t)d * C * HID, f1b + d * HID,
                    nullptr, big, nullptr, M, HID, C, 1, stream);
        launch_gemm(big, f2w + (size_t)d * HID * C, f2b + d * C,
                    carry, carry, (d < 4 ? acc1 : acc0), M, C, HID, 0, stream);
    }

    // output assembly: stack([en0, en1, de0, de1]) as (4,B,C,32,32)
    const size_t slot = (size_t)4 << 20;                  // B*C*1024
    const dim3 tgrid(32, 32, 4);
    // en0: mean of en_feats[:, :4]
    transpose_out_kernel<<<tgrid, blk, 0, stream>>>(
        en, outp, (size_t)8 << 20, 4, (size_t)1 << 20, 0.25f);
    // en1: mean of en_feats[:, 4:]
    transpose_out_kernel<<<tgrid, blk, 0, stream>>>(
        en + ((size_t)4 << 20), outp + slot, (size_t)8 << 20, 4, (size_t)1 << 20, 0.25f);
    // de0: mean of carries from blocks 4..7 (reversed order first half)
    transpose_out_kernel<<<tgrid, blk, 0, stream>>>(
        acc0, outp + 2 * slot, (size_t)1 << 20, 1, 0, 0.25f);
    // de1: mean of carries from blocks 0..3
    transpose_out_kernel<<<tgrid, blk, 0, stream>>>(
        acc1, outp + 3 * slot, (size_t)1 << 20, 1, 0, 0.25f);
}
